// SphericalVectorQuantizer_29008209117616
// MI455X (gfx1250) — matrix operand order repaired and checked by reading, not compiled, not run on hardware
//
#include <hip/hip_runtime.h>

// ---------------------------------------------------------------------------
// Spherical VQ for MI455X (gfx1250, wave32):
//   z_e = rmsnorm(z); cb = rmsnorm(batchnorm(codebook));
//   q = argmin_n (|cb_n|^2 - 2 * z_e . cb_n);  z_q = cb[q]
// Fused bf16-WMMA GEMM + argmin; codebook tiles staged through LDS (shared by
// all 8 waves of a block); streaming outputs use non-temporal stores.
// ---------------------------------------------------------------------------

typedef __bf16 bf16_t;
typedef __bf16 v8bf   __attribute__((ext_vector_type(8)));
typedef __bf16 v16bf  __attribute__((ext_vector_type(16)));
typedef float  v8f    __attribute__((ext_vector_type(8)));

#define NUM_CODES 4096
#define EMBED_DIM 64
#define NROWS     32768          // 8 * 4096
#define EPS_RMS   1.1920929e-07f
#define EPS_BN    1e-5f
#define CHUNK     128            // codes staged per LDS round (16 KB)

// ws layout (float offsets)
#define WS_MEAN   0
#define WS_INVSTD 64
#define WS_CNORM  128                       // 4096 floats
#define WS_CBF32  4224                      // 4096*64 floats
#define WS_BF16_BYTE ((size_t)(4224 + NUM_CODES * EMBED_DIM) * 4)

// out layout (float offsets): (z_q_st, z_e, q, z_q)
#define ZQST_OFF  ((size_t)0)
#define ZE_OFF    ((size_t)NROWS * EMBED_DIM)            // 2097152
#define Q_OFF     ((size_t)2 * NROWS * EMBED_DIM)        // 4194304
#define ZQ_OFF    (Q_OFF + NROWS)                        // 4227072

// ---------------------------------------------------------------------------
// Kernel 1: per-column batch stats of the codebook (biased variance).
// 64 blocks (one per column) x 256 threads.
// ---------------------------------------------------------------------------
__global__ void __launch_bounds__(256) bn_stats_kernel(
    const float* __restrict__ cw, float* __restrict__ ws) {
  __shared__ float s_sum[256];
  __shared__ float s_sq[256];
  const int c = blockIdx.x;
  const int t = threadIdx.x;
  float sum = 0.f, sq = 0.f;
  for (int r = t; r < NUM_CODES; r += 256) {
    float w = cw[(size_t)r * EMBED_DIM + c];
    sum += w;
    sq  += w * w;
  }
  s_sum[t] = sum; s_sq[t] = sq;
  __syncthreads();
  for (int s = 128; s > 0; s >>= 1) {
    if (t < s) { s_sum[t] += s_sum[t + s]; s_sq[t] += s_sq[t + s]; }
    __syncthreads();
  }
  if (t == 0) {
    float mean = s_sum[0] * (1.0f / NUM_CODES);
    float var  = s_sq[0] * (1.0f / NUM_CODES) - mean * mean;
    ws[WS_MEAN + c]   = mean;
    ws[WS_INVSTD + c] = rsqrtf(var + EPS_BN);
  }
}

// ---------------------------------------------------------------------------
// Kernel 2: cb = rmsnorm(bn(codebook)). One wave (32 lanes) per code row,
// 2 elements per lane. Writes cb f32, cb bf16, |cb|^2.
// ---------------------------------------------------------------------------
__global__ void __launch_bounds__(256) build_codebook_kernel(
    const float* __restrict__ cw, const float* __restrict__ gamma,
    const float* __restrict__ beta, float* __restrict__ ws,
    bf16_t* __restrict__ cb_bf) {
  const int tid  = threadIdx.x;
  const int lane = tid & 31;
  const int w    = tid >> 5;
  const int k    = blockIdx.x * 8 + w;          // code row
  const int c0   = lane * 2;

  const float m0 = ws[WS_MEAN + c0],     m1 = ws[WS_MEAN + c0 + 1];
  const float i0 = ws[WS_INVSTD + c0],   i1 = ws[WS_INVSTD + c0 + 1];
  const float g0 = gamma[c0],            g1 = gamma[c0 + 1];
  const float b0 = beta[c0],             b1 = beta[c0 + 1];

  const size_t base = (size_t)k * EMBED_DIM + c0;
  float h0 = (cw[base]     - m0) * i0 * g0 + b0;
  float h1 = (cw[base + 1] - m1) * i1 * g1 + b1;

  float sq = h0 * h0 + h1 * h1;
  #pragma unroll
  for (int mask = 16; mask; mask >>= 1) sq += __shfl_xor(sq, mask, 32);

  const float ms = sq * (1.0f / EMBED_DIM);
  const float r  = rsqrtf(ms + EPS_RMS);
  const float o0 = h0 * r, o1 = h1 * r;

  float* cbf = ws + WS_CBF32;
  cbf[base]     = o0;
  cbf[base + 1] = o1;
  cb_bf[base]     = (bf16_t)o0;
  cb_bf[base + 1] = (bf16_t)o1;
  if (lane == 0) ws[WS_CNORM + k] = sq / (ms + EPS_RMS);  // = sum(cb^2)
}

// ---------------------------------------------------------------------------
// Kernel 3: z_e = rmsnorm(z). One wave per row; writes z_e f32 (output,
// non-temporal) and a bf16 copy for the WMMA A operand.
// ---------------------------------------------------------------------------
__global__ void __launch_bounds__(256) rms_z_kernel(
    const float* __restrict__ z, float* __restrict__ out,
    bf16_t* __restrict__ z_bf) {
  const int tid  = threadIdx.x;
  const int lane = tid & 31;
  const int w    = tid >> 5;
  const int row  = blockIdx.x * 8 + w;
  const size_t base = (size_t)row * EMBED_DIM + lane * 2;

  float x0 = __builtin_nontemporal_load(z + base);       // streamed once
  float x1 = __builtin_nontemporal_load(z + base + 1);
  float sq = x0 * x0 + x1 * x1;
  #pragma unroll
  for (int mask = 16; mask; mask >>= 1) sq += __shfl_xor(sq, mask, 32);

  const float r = rsqrtf(sq * (1.0f / EMBED_DIM) + EPS_RMS);
  const float e0 = x0 * r, e1 = x1 * r;

  __builtin_nontemporal_store(e0, out + ZE_OFF + base);
  __builtin_nontemporal_store(e1, out + ZE_OFF + base + 1);
  z_bf[base]     = (bf16_t)e0;
  z_bf[base + 1] = (bf16_t)e1;
}

// ---------------------------------------------------------------------------
// Kernel 4: fused bf16-WMMA GEMM + argmin + gather.
// 256 threads = 8 waves; each wave owns a 16-row M-tile. The codebook is
// swept in 128-code chunks staged into LDS once per block (8x reuse),
// 16-wide N-tiles, 2 chained v_wmma_f32_16x16x32_bf16 per tile (K = 64).
// ---------------------------------------------------------------------------
__global__ void __launch_bounds__(256) dist_argmin_kernel(
    const bf16_t* __restrict__ zb, const bf16_t* __restrict__ cb_bf,
    const float* __restrict__ ws, float* __restrict__ out) {
  __shared__ float  s_cn[NUM_CODES];                         // 16 KB
  __shared__ __align__(32) bf16_t s_cb[CHUNK * EMBED_DIM];   // 16 KB
  const int tid = threadIdx.x;
  for (int i = tid; i < NUM_CODES; i += 256) s_cn[i] = ws[WS_CNORM + i];

  const int lane = tid & 31;
  const int w    = tid >> 5;
  const int half = lane >> 4;     // 0: lanes 0-15, 1: lanes 16-31
  const int lidx = lane & 15;
  const int m0   = (blockIdx.x * 8 + w) * 16;

  // --- Load A tile (16x64 bf16) in contiguous K order (same as B):
  // lane holds row m = lane&15; element e of chunk c -> K = c*32 + half*16 + e
  v16bf A[2];
  #pragma unroll
  for (int c = 0; c < 2; ++c) {
    const bf16_t* base = zb + (size_t)(m0 + lidx) * EMBED_DIM + c * 32 + half * 16;
    A[c] = *reinterpret_cast<const v16bf*>(base);
  }

  float best[8];
  int   bidx[8];
  #pragma unroll
  for (int v = 0; v < 8; ++v) { best[v] = 3.4e38f; bidx[v] = 0; }

  // Per-thread staging slots: 16384 B / 256 threads = two v16bf each.
  v16bf* s_cb_v = reinterpret_cast<v16bf*>(s_cb);

  for (int chunk = 0; chunk < NUM_CODES / CHUNK; ++chunk) {
    // Cooperative stage of 128 codes (16 KB) into LDS.
    const v16bf* src = reinterpret_cast<const v16bf*>(
        cb_bf + (size_t)chunk * CHUNK * EMBED_DIM);
    s_cb_v[tid * 2]     = src[tid * 2];
    s_cb_v[tid * 2 + 1] = src[tid * 2 + 1];
    __syncthreads();

    #pragma unroll 2
    for (int t = 0; t < CHUNK / 16; ++t) {
      const int nl = t * 16 + lidx;            // code index within chunk
      const int n  = chunk * CHUNK + nl;       // global code index
      // B layout (32x16 bf16, K x N): lane -> N = lane&15, element e -> K = half*16 + e
      const bf16_t* bbase = s_cb + (size_t)nl * EMBED_DIM + half * 16;
      v16bf B0 = *reinterpret_cast<const v16bf*>(bbase);        // K = 0..31
      v16bf B1 = *reinterpret_cast<const v16bf*>(bbase + 32);   // K = 32..63

      v8f acc = {};
      acc = __builtin_amdgcn_wmma_f32_16x16x32_bf16(false, A[0], false, B0,
                                                    (short)0, acc, false, false);
      acc = __builtin_amdgcn_wmma_f32_16x16x32_bf16(false, A[1], false, B1,
                                                    (short)0, acc, false, false);

      const float cn = s_cn[n];
      #pragma unroll
      for (int v = 0; v < 8; ++v) {
        // D layout: lane holds n = lane&15, m = v + half*8 for VGPR v
        float d = cn - 2.0f * acc[v];
        if (d < best[v]) { best[v] = d; bidx[v] = n; }  // ascending n => first-min kept
      }
    }
    __syncthreads();   // protect s_cb before next stage overwrites it
  }

  // Min-reduce across the 16 lanes of each half (tie-break: lowest index).
  #pragma unroll
  for (int v = 0; v < 8; ++v) {
    #pragma unroll
    for (int mask = 8; mask; mask >>= 1) {
      float od = __shfl_xor(best[v], mask, 32);
      int   oi = __shfl_xor(bidx[v], mask, 32);
      if (od < best[v] || (od == best[v] && oi < bidx[v])) {
        best[v] = od; bidx[v] = oi;
      }
    }
  }

  // Gather winning codes and write z_q_st, q, z_q (z_q_st == z_q numerically).
  const float* cbf = ws + WS_CBF32;
  #pragma unroll
  for (int h = 0; h < 2; ++h) {
    #pragma unroll
    for (int v = 0; v < 8; ++v) {
      const int idx = __shfl(bidx[v], h * 16, 32);
      const int gm  = m0 + h * 8 + v;
      const size_t rb = (size_t)gm * EMBED_DIM + lane * 2;
      const float a0 = cbf[(size_t)idx * EMBED_DIM + lane * 2];
      const float a1 = cbf[(size_t)idx * EMBED_DIM + lane * 2 + 1];
      __builtin_nontemporal_store(a0, out + ZQST_OFF + rb);
      __builtin_nontemporal_store(a1, out + ZQST_OFF + rb + 1);
      __builtin_nontemporal_store(a0, out + ZQ_OFF + rb);
      __builtin_nontemporal_store(a1, out + ZQ_OFF + rb + 1);
      if (lane == 0) __builtin_nontemporal_store((float)idx, out + Q_OFF + gm);
    }
  }
}

// ---------------------------------------------------------------------------
extern "C" void kernel_launch(void* const* d_in, const int* in_sizes, int n_in,
                              void* d_out, int out_size, void* d_ws, size_t ws_size,
                              hipStream_t stream) {
  const float* z   = (const float*)d_in[0];   // [8,4096,64]
  const float* cw  = (const float*)d_in[1];   // [4096,64]
  const float* bnw = (const float*)d_in[2];   // [64]
  const float* bnb = (const float*)d_in[3];   // [64]
  float* out = (float*)d_out;
  float* ws  = (float*)d_ws;

  bf16_t* cb_bf = (bf16_t*)((char*)d_ws + WS_BF16_BYTE);
  bf16_t* z_bf  = cb_bf + (size_t)NUM_CODES * EMBED_DIM;

  bn_stats_kernel<<<EMBED_DIM, 256, 0, stream>>>(cw, ws);
  build_codebook_kernel<<<NUM_CODES / 8, 256, 0, stream>>>(cw, bnw, bnb, ws, cb_bf);
  rms_z_kernel<<<NROWS / 8, 256, 0, stream>>>(z, out, z_bf);
  dist_argmin_kernel<<<NROWS / 128, 256, 0, stream>>>(z_bf, cb_bf, ws, out);
}
